// CausalMultiGranularAttention_12378095747731
// MI455X (gfx1250) — matrix operand order repaired and checked by reading, not compiled, not run on hardware
//
#include <hip/hip_runtime.h>
#include <hip/hip_bf16.h>
#include <cstdint>
#include <cstddef>

// ---------------- Problem constants (compile-time) ----------------
constexpr int Bc  = 2;
constexpr int Sc  = 2048;
constexpr int Hc  = 1024;
constexpr int NHc = 16;
constexpr int HDc = 64;
constexpr int Mc  = Bc * Sc;           // 4096 rows for projections

// ---------------- WMMA / descriptor types ----------------
typedef __attribute__((ext_vector_type(16))) __bf16        v16bf;
typedef __attribute__((ext_vector_type(8)))  float         v8f;
typedef __attribute__((ext_vector_type(4)))  unsigned int  v4u;
typedef __attribute__((ext_vector_type(8)))  unsigned int  v8u;

union Frag16 {
    v16bf v;
    uint4 q[2];
    unsigned short s[16];
    unsigned int   u[8];
};

__device__ __forceinline__ unsigned short f2bf(float f) {
    union { float f; unsigned int u; } c; c.f = f;
    unsigned int u = c.u;
    unsigned int r = u + 0x7FFFu + ((u >> 16) & 1u);   // round-to-nearest-even
    return (unsigned short)(r >> 16);
}

__device__ __forceinline__ v16bf load_frag_2x8(const unsigned short* p0,
                                               const unsigned short* p1) {
    Frag16 f;
    f.q[0] = *reinterpret_cast<const uint4*>(p0);
    f.q[1] = *reinterpret_cast<const uint4*>(p1);
    return f.v;
}

// ---- gfx1250 async global->LDS copies (ASYNCcnt) ----
__device__ __forceinline__ void async_copy_b128(const void* g, void* lds) {
    unsigned laddr = (unsigned)(uintptr_t)lds;                 // low 32b = LDS offset
    unsigned long long ga = (unsigned long long)(uintptr_t)g;
    asm volatile("global_load_async_to_lds_b128 %0, %1, off"
                 :: "v"(laddr), "v"(ga) : "memory");
}
__device__ __forceinline__ void async_copy_b64(const void* g, void* lds) {
    unsigned laddr = (unsigned)(uintptr_t)lds;
    unsigned long long ga = (unsigned long long)(uintptr_t)g;
    asm volatile("global_load_async_to_lds_b64 %0, %1, off"
                 :: "v"(laddr), "v"(ga) : "memory");
}
__device__ __forceinline__ void async_wait0() {
    asm volatile("s_wait_asynccnt 0x0" ::: "memory");
}

// =====================================================================
// Kernel 1/5: tiled GEMM  Y = A @ W^T + bias   (M=4096, N=1024, K=1024)
//   A_IS_BF16=false : A is f32 (x), converted to bf16 on the fly
//   A_IS_BF16=true  : A is bf16 (ctx), staged via async-to-LDS
//   QKV_OUT=true    : output bf16 scattered to [B,NH,S,HD], scaled
//   QKV_OUT=false   : output f32 row-major [M,H]  (final projection)
// Block: 256 thr (8 waves), tile 128(M) x 64(N), K-step 32.
// =====================================================================
template<bool A_IS_BF16, bool QKV_OUT>
__global__ __launch_bounds__(256)
void gemm_wmma_kernel(const void* __restrict__ Ain,
                      const float* __restrict__ W,
                      const float* __restrict__ bias,
                      void* __restrict__ out,
                      float scale)
{
    __shared__ unsigned short Asm[128 * 32];   // 8 KB bf16
    __shared__ unsigned short Bsm[64 * 32];    // 4 KB bf16

    const int tid  = threadIdx.x;
    const int wid  = tid >> 5;
    const int lane = tid & 31;
    const int mblock = blockIdx.x * 128;
    const int nblock = blockIdx.y * 64;

    v8f acc[4];
    #pragma unroll
    for (int i = 0; i < 4; ++i) acc[i] = (v8f){0.f,0.f,0.f,0.f,0.f,0.f,0.f,0.f};

    const int arow  = tid >> 1;          // 0..127
    const int acolb = (tid & 1) * 16;    // 0 or 16
    const int brow  = tid >> 2;          // 0..63
    const int bcolb = (tid & 3) * 8;     // 0,8,16,24

    for (int kt = 0; kt < Hc; kt += 32) {
        // ---- stage A tile (128x32) ----
        if (A_IS_BF16) {
            const unsigned short* As =
                (const unsigned short*)Ain + (size_t)(mblock + arow) * Hc + kt + acolb;
            async_copy_b128(As,     &Asm[arow * 32 + acolb]);
            async_copy_b128(As + 8, &Asm[arow * 32 + acolb + 8]);
            if (kt + 32 < Hc) __builtin_prefetch(As + 32, 0, 0);
        } else {
            const float* Af = (const float*)Ain + (size_t)(mblock + arow) * Hc + kt + acolb;
            unsigned int* dst = reinterpret_cast<unsigned int*>(&Asm[arow * 32 + acolb]);
            #pragma unroll
            for (int i2 = 0; i2 < 8; ++i2) {
                float f0 = Af[2 * i2], f1 = Af[2 * i2 + 1];
                dst[i2] = (unsigned int)f2bf(f0) | ((unsigned int)f2bf(f1) << 16);
            }
            if (kt + 32 < Hc) __builtin_prefetch(Af + 32, 0, 0);
        }
        // ---- stage B tile (64x32): rows of W (columns of W^T) ----
        {
            const float* Wf = W + (size_t)(nblock + brow) * Hc + kt + bcolb;
            unsigned int* dst = reinterpret_cast<unsigned int*>(&Bsm[brow * 32 + bcolb]);
            #pragma unroll
            for (int i2 = 0; i2 < 4; ++i2) {
                float f0 = Wf[2 * i2], f1 = Wf[2 * i2 + 1];
                dst[i2] = (unsigned int)f2bf(f0) | ((unsigned int)f2bf(f1) << 16);
            }
            if (kt + 32 < Hc) __builtin_prefetch(Wf + 32, 0, 0);
        }
        if (A_IS_BF16) async_wait0();
        __syncthreads();

        // ---- A fragment: 16x32 bf16 (ISA layout) ----
        const int afrow = wid * 16 + (lane & 15);
        const int akoff = (lane < 16) ? 0 : 8;
        v16bf afrag = load_frag_2x8(&Asm[afrow * 32 + akoff],
                                    &Asm[afrow * 32 + akoff + 16]);
        // ---- 4 N tiles ----
        #pragma unroll
        for (int nt = 0; nt < 4; ++nt) {
            const int bfrow = nt * 16 + (lane & 15);
            const int bkoff = (lane < 16) ? 0 : 8;
            v16bf bfrag = load_frag_2x8(&Bsm[bfrow * 32 + bkoff],
                                        &Bsm[bfrow * 32 + bkoff + 16]);
            acc[nt] = __builtin_amdgcn_wmma_f32_16x16x32_bf16(
                false, afrag, false, bfrag, (short)0, acc[nt], false, false);
        }
        __syncthreads();
    }

    // ---- epilogue ----
    const int mrow_lo = mblock + wid * 16 + ((lane < 16) ? 0 : 8);
    #pragma unroll
    for (int nt = 0; nt < 4; ++nt) {
        const int n = nblock + nt * 16 + (lane & 15);
        const float bv = bias[n];
        #pragma unroll
        for (int r = 0; r < 8; ++r) {
            const int m = mrow_lo + r;
            float val = (acc[nt][r] + bv) * scale;
            if (QKV_OUT) {
                // scatter to [B, NH, S, HD] bf16
                const int bb = m / Sc, s = m % Sc;
                const int h = n / HDc, d = n % HDc;
                ((unsigned short*)out)[(((size_t)bb * NHc + h) * Sc + s) * HDc + d] = f2bf(val);
            } else {
                ((float*)out)[(size_t)m * Hc + n] = val;
            }
        }
    }
}

// =====================================================================
// Kernel 2/5: masked scores.  grid (S/16, NH, B), block 256 (8 waves).
// Q tile staged via async global->LDS; wave w does col tiles
// jt = w, w+8, ... <= qb (causal). Raw scores -> attn (f32), -inf above.
// =====================================================================
__global__ __launch_bounds__(256)
void scores_wmma_kernel(const unsigned short* __restrict__ q,
                        const unsigned short* __restrict__ k,
                        float* __restrict__ attn)
{
    __shared__ unsigned short Qs[16 * HDc];   // 2 KB

    const int tid  = threadIdx.x;
    const int wid  = tid >> 5;
    const int lane = tid & 31;
    const int qb = blockIdx.x;                 // query tile index (16 rows)
    const int h  = blockIdx.y;
    const int b  = blockIdx.z;
    const size_t headoff = ((size_t)b * NHc + h) * Sc * HDc;   // q/k element offset
    const size_t attnrow0 = (((size_t)b * NHc + h) * Sc + (size_t)qb * 16) * Sc;

    // stage Q tile: 16*64 bf16 = 2048B -> 8 bytes/thread via async copy
    {
        const unsigned short* qsrc = q + headoff + (size_t)qb * 16 * HDc + tid * 4;
        async_copy_b64(qsrc, &Qs[tid * 4]);
        async_wait0();
    }
    __syncthreads();

    // A fragments for K-dim 0..31 and 32..63
    const int qrow = lane & 15;
    const int koff = (lane < 16) ? 0 : 8;
    v16bf a0 = load_frag_2x8(&Qs[qrow * HDc + koff],      &Qs[qrow * HDc + koff + 16]);
    v16bf a1 = load_frag_2x8(&Qs[qrow * HDc + 32 + koff], &Qs[qrow * HDc + 32 + koff + 16]);

    const int ilocal = ((lane < 16) ? 0 : 8);
    for (int jt = wid; jt <= qb; jt += 8) {
        const unsigned short* kp = k + headoff + (size_t)(jt * 16 + (lane & 15)) * HDc;
        const int doff = (lane < 16) ? 0 : 8;
        v16bf b0 = load_frag_2x8(kp + doff,      kp + doff + 16);
        v16bf b1 = load_frag_2x8(kp + 32 + doff, kp + 32 + doff + 16);

        v8f acc = (v8f){0.f,0.f,0.f,0.f,0.f,0.f,0.f,0.f};
        acc = __builtin_amdgcn_wmma_f32_16x16x32_bf16(false, a0, false, b0, (short)0, acc, false, false);
        acc = __builtin_amdgcn_wmma_f32_16x16x32_bf16(false, a1, false, b1, (short)0, acc, false, false);

        if (jt == qb) {   // diagonal tile: mask j_local > i_local
            #pragma unroll
            for (int r = 0; r < 8; ++r)
                if ((lane & 15) > ilocal + r) acc[r] = -__builtin_inff();
        }
        #pragma unroll
        for (int r = 0; r < 8; ++r)
            attn[attnrow0 + (size_t)(ilocal + r) * Sc + jt * 16 + (lane & 15)] = acc[r];
    }

    // fill -inf strictly above the diagonal block (cols >= (qb+1)*16)
    // row-outer / strided-col loops: no integer division.
    const int cstart = (qb + 1) * 16;
    const float4 ninf = make_float4(-__builtin_inff(), -__builtin_inff(),
                                    -__builtin_inff(), -__builtin_inff());
    for (int r = 0; r < 16; ++r) {
        float* rowp = attn + attnrow0 + (size_t)r * Sc;
        for (int c = cstart + tid * 4; c < Sc; c += 1024)
            *reinterpret_cast<float4*>(rowp + c) = ninf;
    }
}

// =====================================================================
// Kernel 3/5: row softmax over attn.  grid B*NH*S blocks, 256 threads.
// 8 f32 per thread held in registers: one read + one write of attn.
// =====================================================================
__global__ __launch_bounds__(256)
void softmax_kernel(float* __restrict__ attn)
{
    __shared__ float redmax[8];
    __shared__ float redsum[8];

    const int tid  = threadIdx.x;
    const int wid  = tid >> 5;
    const int lane = tid & 31;
    float* p = attn + (size_t)blockIdx.x * Sc + tid * 8;

    float4 v0 = reinterpret_cast<float4*>(p)[0];
    float4 v1 = reinterpret_cast<float4*>(p)[1];

    float m = fmaxf(fmaxf(fmaxf(v0.x, v0.y), fmaxf(v0.z, v0.w)),
                    fmaxf(fmaxf(v1.x, v1.y), fmaxf(v1.z, v1.w)));
    #pragma unroll
    for (int off = 16; off >= 1; off >>= 1) m = fmaxf(m, __shfl_xor(m, off, 32));
    if (lane == 0) redmax[wid] = m;
    __syncthreads();
    float M = redmax[0];
    #pragma unroll
    for (int i = 1; i < 8; ++i) M = fmaxf(M, redmax[i]);

    v0.x = __expf(v0.x - M); v0.y = __expf(v0.y - M);
    v0.z = __expf(v0.z - M); v0.w = __expf(v0.w - M);
    v1.x = __expf(v1.x - M); v1.y = __expf(v1.y - M);
    v1.z = __expf(v1.z - M); v1.w = __expf(v1.w - M);

    float s = (v0.x + v0.y + v0.z + v0.w) + (v1.x + v1.y + v1.z + v1.w);
    #pragma unroll
    for (int off = 16; off >= 1; off >>= 1) s += __shfl_xor(s, off, 32);
    if (lane == 0) redsum[wid] = s;
    __syncthreads();
    float T = 0.f;
    #pragma unroll
    for (int i = 0; i < 8; ++i) T += redsum[i];
    const float inv = 1.0f / T;

    v0.x *= inv; v0.y *= inv; v0.z *= inv; v0.w *= inv;
    v1.x *= inv; v1.y *= inv; v1.z *= inv; v1.w *= inv;
    reinterpret_cast<float4*>(p)[0] = v0;
    reinterpret_cast<float4*>(p)[1] = v1;
}

// =====================================================================
// Kernel 4/5: ctx = attn @ V per head.  grid (S/64, NH, B), block 256.
// V tile (32x64 bf16, row-major) DMA'd into LDS by the Tensor Data
// Mover (2D D#: data_size=2B, tensor_dim0=64, dim0_stride=64, tensor
// rows=S, tile 64x32), then transposed LDS->LDS for B fragments.
// Causal truncation: K-loop only over j < (qb64+1)*64.
// =====================================================================
__global__ __launch_bounds__(256)
void ctx_wmma_kernel(const float* __restrict__ attn,
                     const unsigned short* __restrict__ v,
                     unsigned short* __restrict__ ctx)
{
    __shared__ unsigned short Vraw[32 * HDc];  // row-major V tile, 4 KB
    __shared__ unsigned short VT[HDc * 32];    // transposed [d][k], 4 KB

    const int tid  = threadIdx.x;
    const int wid  = tid >> 5;
    const int lane = tid & 31;
    const int qb64 = blockIdx.x;
    const int h    = blockIdx.y;
    const int b    = blockIdx.z;
    const size_t headoff = ((size_t)b * NHc + h) * Sc * HDc;
    const size_t attnhead = ((size_t)b * NHc + h) * Sc;

    const int mt  = wid >> 1;           // 0..3
    const int ntb = (wid & 1) * 2;      // 0 or 2

    v8f acc[2];
    acc[0] = (v8f){0.f,0.f,0.f,0.f,0.f,0.f,0.f,0.f};
    acc[1] = acc[0];

    const int arow = qb64 * 64 + mt * 16 + (lane & 15);
    const int kend = (qb64 + 1) * 64;

    const int vr = tid >> 3;            // 0..31 (V row within K-step)
    const int db = (tid & 7) * 8;       // d base

    for (int j = 0; j < kend; j += 32) {
        // ---- TDM: DMA V[j..j+31][0..63] (bf16 row-major) into Vraw ----
        if (wid == 0) {
            const unsigned lds_off = (unsigned)(uintptr_t)(void*)&Vraw[0];
            const unsigned long long ga =
                (unsigned long long)(uintptr_t)(const void*)(v + headoff + (size_t)j * HDc);
            v4u g0;
            g0.x = 1u;                                        // count=1 (valid user D#)
            g0.y = lds_off;                                   // lds_addr
            g0.z = (unsigned)ga;                              // global_addr[31:0]
            g0.w = ((unsigned)(ga >> 32) & 0x01FFFFFFu) | (2u << 30); // addr[56:32], type=2
            v8u g1;
            g1[0] = 0x00010000u;                              // data_size=1 -> 2 bytes
            g1[1] = (unsigned)HDc << 16;                      // tensor_dim0 = 64
            g1[2] = (unsigned)Sc << 16;                       // tensor_dim1 = 2048
            g1[3] = (unsigned)HDc << 16;                      // tile_dim0 = 64
            g1[4] = 32u;                                      // tile_dim1 = 32
            g1[5] = (unsigned)HDc;                            // tensor_dim0_stride = 64
            g1[6] = 0u;
            g1[7] = 0u;
            asm volatile("tensor_load_to_lds %0, %1" :: "s"(g0), "s"(g1) : "memory");
            __builtin_amdgcn_s_wait_tensorcnt((unsigned short)0);
        }
        __syncthreads();

        // ---- transpose Vraw -> VT (LDS to LDS) ----
        {
            union { uint4 q; unsigned short s[8]; } u;
            u.q = *reinterpret_cast<const uint4*>(&Vraw[vr * HDc + db]);
            #pragma unroll
            for (int i = 0; i < 8; ++i) VT[(db + i) * 32 + vr] = u.s[i];
        }
        __syncthreads();

        // ---- A fragment from attn (f32 -> bf16 on the fly) ----
        const float* pa = attn + (attnhead + arow) * Sc + j;
        const int koff = (lane < 16) ? 0 : 8;
        Frag16 af;
        {
            float4 f0 = *reinterpret_cast<const float4*>(pa + koff);
            float4 f1 = *reinterpret_cast<const float4*>(pa + koff + 4);
            float4 f2 = *reinterpret_cast<const float4*>(pa + koff + 16);
            float4 f3 = *reinterpret_cast<const float4*>(pa + koff + 20);
            af.s[0]=f2bf(f0.x); af.s[1]=f2bf(f0.y); af.s[2]=f2bf(f0.z); af.s[3]=f2bf(f0.w);
            af.s[4]=f2bf(f1.x); af.s[5]=f2bf(f1.y); af.s[6]=f2bf(f1.z); af.s[7]=f2bf(f1.w);
            af.s[8]=f2bf(f2.x); af.s[9]=f2bf(f2.y); af.s[10]=f2bf(f2.z); af.s[11]=f2bf(f2.w);
            af.s[12]=f2bf(f3.x); af.s[13]=f2bf(f3.y); af.s[14]=f2bf(f3.z); af.s[15]=f2bf(f3.w);
        }
        #pragma unroll
        for (int t2 = 0; t2 < 2; ++t2) {
            const int drow = (ntb + t2) * 16 + (lane & 15);
            const int koff2 = (lane < 16) ? 0 : 8;
            v16bf bf = load_frag_2x8(&VT[drow * 32 + koff2], &VT[drow * 32 + koff2 + 16]);
            acc[t2] = __builtin_amdgcn_wmma_f32_16x16x32_bf16(
                false, af.v, false, bf, (short)0, acc[t2], false, false);
        }
        __syncthreads();
    }

    // epilogue: ctx bf16, [B,S,H] layout
    const int srow_lo = qb64 * 64 + mt * 16 + ((lane < 16) ? 0 : 8);
    #pragma unroll
    for (int t2 = 0; t2 < 2; ++t2) {
        const int d = (ntb + t2) * 16 + (lane & 15);
        #pragma unroll
        for (int r = 0; r < 8; ++r) {
            const int s = srow_lo + r;
            ctx[((size_t)b * Sc + s) * Hc + h * HDc + d] = f2bf(acc[t2][r]);
        }
    }
}

// =====================================================================
// Host-side launcher
// =====================================================================
extern "C" void kernel_launch(void* const* d_in, const int* in_sizes, int n_in,
                              void* d_out, int out_size, void* d_ws, size_t ws_size,
                              hipStream_t stream) {
    (void)in_sizes; (void)n_in; (void)out_size; (void)ws_size;

    const float* x  = (const float*)d_in[0];
    const float* Wq = (const float*)d_in[1];
    const float* bq = (const float*)d_in[2];
    const float* Wk = (const float*)d_in[3];
    const float* bk = (const float*)d_in[4];
    const float* Wv = (const float*)d_in[5];
    const float* bv = (const float*)d_in[6];
    const float* Wp = (const float*)d_in[7];
    const float* bp = (const float*)d_in[8];

    // workspace: q, k, v, ctx as bf16 (each 4096*1024 elements = 8 MB)
    unsigned short* qb_ = (unsigned short*)d_ws;
    unsigned short* kb_ = qb_ + (size_t)Mc * Hc;
    unsigned short* vb_ = kb_ + (size_t)Mc * Hc;
    unsigned short* cb_ = vb_ + (size_t)Mc * Hc;

    float* outp = (float*)d_out;                       // [B,S,H]
    float* attn = outp + (size_t)Mc * Hc;              // [B,NH,S,S]

    const dim3 gblk(256);
    const dim3 ggrid(Mc / 128, Hc / 64);               // (32,16)
    const float qscale = 0.125f;                       // 1/sqrt(64)

    gemm_wmma_kernel<false, true><<<ggrid, gblk, 0, stream>>>(x, Wq, bq, qb_, qscale);
    gemm_wmma_kernel<false, true><<<ggrid, gblk, 0, stream>>>(x, Wk, bk, kb_, 1.0f);
    gemm_wmma_kernel<false, true><<<ggrid, gblk, 0, stream>>>(x, Wv, bv, vb_, 1.0f);

    scores_wmma_kernel<<<dim3(Sc / 16, NHc, Bc), gblk, 0, stream>>>(qb_, kb_, attn);

    softmax_kernel<<<dim3(Bc * NHc * Sc), gblk, 0, stream>>>(attn);

    ctx_wmma_kernel<<<dim3(Sc / 64, NHc, Bc), gblk, 0, stream>>>(attn, vb_, cb_);

    gemm_wmma_kernel<true, false><<<ggrid, gblk, 0, stream>>>(cb_, Wp, bp, outp, 1.0f);
}
